// EdgeEmbedding_34797825032831
// MI455X (gfx1250) — matrix operand order repaired and checked by reading, not compiled, not run on hardware
//
#include <hip/hip_runtime.h>
#include <cstdint>

typedef __bf16 v16bf __attribute__((ext_vector_type(16)));
typedef float  v8f   __attribute__((ext_vector_type(8)));
typedef int    ai4   __attribute__((ext_vector_type(4)));

#if defined(__AMDGCN__) && __has_builtin(__builtin_amdgcn_global_load_async_to_lds_b128) && \
    __has_builtin(__builtin_amdgcn_s_wait_asynccnt)
#define USE_ASYNC_LDS 1
#else
#define USE_ASYNC_LDS 0
#endif

constexpr int ATOMF     = 256;
constexpr int EDGEF     = 128;
constexpr int OUTF      = 256;
constexpr int INF       = 2 * ATOMF + EDGEF;  // 640
constexpr int KSTEPS    = INF / 32;           // 20 K-steps of 32
constexpr int WAVES     = 8;
constexpr int BLK_EDGES = WAVES * 16;         // 128 edges per block
constexpr int BSTEP_DW  = 32 * OUTF / 2;      // 4096 dwords of packed bf16 W per K-step

__device__ __forceinline__ uint32_t pack2bf(float a, float b) {
  uint32_t ua = __builtin_bit_cast(uint32_t, a);
  uint32_t ub = __builtin_bit_cast(uint32_t, b);
  ua = (ua + 0x7FFFu + ((ua >> 16) & 1u)) >> 16;  // RNE f32 -> bf16
  ub = (ub + 0x7FFFu + ((ub >> 16) & 1u)) >> 16;
  return (ua & 0xFFFFu) | (ub << 16);
}

// Repack W (fp32, [k][n] row-major, 640x256) into bf16 B-fragment layout:
// dword index = ((s*16 + t)*8 + i)*32 + lane, lane = n + 16*h,
// payload = { W[kk][16t+n], W[kk+1][16t+n] } as a bf16 pair, kk = 32s+16(i>>2)+8h+2(i&3).
__global__ void repack_W_kernel(const float* __restrict__ W, uint32_t* __restrict__ Wf) {
  int idx = blockIdx.x * blockDim.x + threadIdx.x;
  if (idx >= KSTEPS * 16 * 8 * 32) return;
  int lane = idx & 31;
  int i    = (idx >> 5) & 7;
  int t    = (idx >> 8) & 15;
  int s    =  idx >> 12;
  int n    = lane & 15;
  int hh   = lane >> 4;
  int k    = 32 * s + 16 * (i >> 2) + 8 * hh + 2 * (i & 3);
  int col  = 16 * t + n;
  Wf[idx] = pack2bf(W[k * OUTF + col], W[(k + 1) * OUTF + col]);
}

__global__ __launch_bounds__(256) void edge_mlp_kernel(
    const float* __restrict__ h, const float* __restrict__ m_rbf,
    const int* __restrict__ idnb_a, const int* __restrict__ idnb_c,
    const uint32_t* __restrict__ Wf, float* __restrict__ out) {
  __shared__ __align__(16) uint32_t Bsh[2 * BSTEP_DW];  // 32 KB double buffer

  const int tid  = threadIdx.x;
  const int lane = tid & 31;
  const int wave = tid >> 5;
  const int n    = lane & 15;  // A-row within wave tile / N-col within subtile
  const int hh   = lane >> 4;  // lane half selects K half

  const int edgeBase = blockIdx.x * BLK_EDGES + wave * 16;
  const int myEdge   = edgeBase + n;  // this lane's A-matrix row
  const float* pA = h     + (size_t)idnb_a[myEdge] * ATOMF;
  const float* pC = h     + (size_t)idnb_c[myEdge] * ATOMF;
  const float* pR = m_rbf + (size_t)myEdge * EDGEF;

  v8f acc[16];
  const v8f vzero = {0.f, 0.f, 0.f, 0.f, 0.f, 0.f, 0.f, 0.f};
#pragma unroll
  for (int t = 0; t < 16; ++t) acc[t] = vzero;

  float2 areg[8];  // raw f32 A staging for next K-step
  auto loadA = [&](int s) {
    const int kb = 32 * s;  // a K-step never straddles the concat boundaries
    const float* src = (kb < ATOMF)     ? (pA + kb)
                     : (kb < 2 * ATOMF) ? (pC + (kb - ATOMF))
                                        : (pR + (kb - 2 * ATOMF));
    const int o = 8 * hh;
#pragma unroll
    for (int i = 0; i < 4; ++i) {
      areg[i]     = *(const float2*)(src + o + 2 * i);       // K = 8h+2i
      areg[4 + i] = *(const float2*)(src + 16 + o + 2 * i);  // K = 16+8h+2i
    }
  };

#if USE_ASYNC_LDS
  // Async global->LDS copy of one 16KB B K-step tile (ASYNCcnt-tracked).
  // Offsets must be frontend-constant immediates, so unrolled by hand.
  auto copyB = [&](int s, int buf) {
    __attribute__((address_space(1))) ai4* gsrc =
        (__attribute__((address_space(1))) ai4*)(Wf + (size_t)s * BSTEP_DW + tid * 4);
    __attribute__((address_space(3))) ai4* ldst =
        (__attribute__((address_space(3))) ai4*)(Bsh + buf * BSTEP_DW + tid * 4);
    // Same IOFFSET applies to both the global and LDS address (ISA Ch10 async).
    __builtin_amdgcn_global_load_async_to_lds_b128(gsrc, ldst, 0,     0);
    __builtin_amdgcn_global_load_async_to_lds_b128(gsrc, ldst, 4096,  0);
    __builtin_amdgcn_global_load_async_to_lds_b128(gsrc, ldst, 8192,  0);
    __builtin_amdgcn_global_load_async_to_lds_b128(gsrc, ldst, 12288, 0);
  };
  copyB(0, 0);
#else
  uint4 breg[4];
  auto loadB = [&](int s) {
    const uint4* g = (const uint4*)(Wf + (size_t)s * BSTEP_DW);
#pragma unroll
    for (int j = 0; j < 4; ++j) breg[j] = g[tid + 256 * j];
  };
  loadB(0);
#endif
  loadA(0);

  union BF { v16bf v; uint32_t u[8]; };

  for (int s = 0; s < KSTEPS; ++s) {
    // Pack the staged A tile into a bf16 WMMA A-fragment (hardware cvt, RNE).
    v16bf af;
#pragma unroll
    for (int i = 0; i < 8; ++i) {
      af[2 * i]     = (__bf16)areg[i].x;
      af[2 * i + 1] = (__bf16)areg[i].y;
    }

#if USE_ASYNC_LDS
    __builtin_amdgcn_s_wait_asynccnt(0);  // my copy(s) landed in LDS
    __syncthreads();                      // everyone's copy(s) landed
    if (s + 1 < KSTEPS) { copyB(s + 1, (s + 1) & 1); loadA(s + 1); }
#else
    uint32_t* dst = Bsh + (s & 1) * BSTEP_DW;
#pragma unroll
    for (int j = 0; j < 4; ++j) ((uint4*)dst)[tid + 256 * j] = breg[j];
    __syncthreads();
    if (s + 1 < KSTEPS) { loadB(s + 1); loadA(s + 1); }
#endif

    // 16 WMMAs with a register-double-buffered B fragment so each v_wmma
    // only waits on LDS loads issued one fragment ahead.
    const uint32_t* bbase = Bsh + (s & 1) * BSTEP_DW + lane;
    BF bf[2];
#pragma unroll
    for (int i = 0; i < 8; ++i) bf[0].u[i] = bbase[i * 32];
#pragma unroll
    for (int t = 0; t < 16; ++t) {
      if (t + 1 < 16) {
#pragma unroll
        for (int i = 0; i < 8; ++i) bf[(t + 1) & 1].u[i] = bbase[(t + 1) * 256 + i * 32];
      }
      acc[t] = __builtin_amdgcn_wmma_f32_16x16x32_bf16(
          false, af, false, bf[t & 1].v, (short)0, acc[t], false, false);
    }
  }

  // Epilogue: SiLU + nontemporal store (C/D layout: VGPR r -> M=r+8h, lane -> N=16t+n).
#pragma unroll
  for (int t = 0; t < 16; ++t) {
#pragma unroll
    for (int r = 0; r < 8; ++r) {
      float x = acc[t][r];
      float y = x / (1.0f + __expf(-x));
      __builtin_nontemporal_store(y, &out[(size_t)(edgeBase + r + 8 * hh) * OUTF + 16 * t + n]);
    }
  }
}

extern "C" void kernel_launch(void* const* d_in, const int* in_sizes, int n_in,
                              void* d_out, int out_size, void* d_ws, size_t ws_size,
                              hipStream_t stream) {
  (void)n_in; (void)out_size; (void)ws_size;
  const float* h      = (const float*)d_in[0];
  const float* m_rbf  = (const float*)d_in[1];
  const int*   idnb_a = (const int*)d_in[2];
  const int*   idnb_c = (const int*)d_in[3];
  const float* W      = (const float*)d_in[4];
  float*       out    = (float*)d_out;
  uint32_t*    Wf     = (uint32_t*)d_ws;  // 327,680 bytes of repacked bf16 W

  const int nEdges = in_sizes[2];

  const int repackElems = KSTEPS * 16 * 8 * 32;
  repack_W_kernel<<<(repackElems + 255) / 256, 256, 0, stream>>>(W, Wf);
  edge_mlp_kernel<<<nEdges / BLK_EDGES, 256, 0, stream>>>(h, m_rbf, idnb_a, idnb_c, Wf, out);
}
